// LSTMCell_31129922961925
// MI455X (gfx1250) — matrix operand order repaired and checked by reading, not compiled, not run on hardware
//
#include <hip/hip_runtime.h>
#include <hip/hip_bf16.h>

typedef __attribute__((ext_vector_type(16))) __bf16 v16bf;
typedef __attribute__((ext_vector_type(8)))  float  v8f;
typedef __attribute__((ext_vector_type(4)))  int    v4i;

#define B_   4096
#define IN_  2048
#define H_   2048
#define K_   (IN_ + H_)   /* 4096: concatenated [x | h] contraction dim */
#define BK   64           /* K per outer step (2 WMMA substeps of 32) */
#define BM   64           /* rows per workgroup */
#define BN   128          /* cols per workgroup (8 waves x 16) */
#define BKP  72           /* BK + 8 pad (bf16 elems), keeps 16B alignment */

#if defined(__AMDGCN__) && __has_builtin(__builtin_amdgcn_global_load_async_to_lds_b128)
#define USE_ASYNC_LDS 1
#else
#define USE_ASYNC_LDS 0
#endif

#if USE_ASYNC_LDS
typedef __attribute__((address_space(1))) v4i gv4i;
typedef __attribute__((address_space(3))) v4i lv4i;

__device__ __forceinline__ void async_cp16(const __bf16* g, __bf16* l) {
    __builtin_amdgcn_global_load_async_to_lds_b128((gv4i*)g, (lv4i*)l, 0, 0);
}
__device__ __forceinline__ void async_wait0() {
#if __has_builtin(__builtin_amdgcn_s_wait_asynccnt)
    __builtin_amdgcn_s_wait_asynccnt(0);
#else
    asm volatile("s_wait_asynccnt 0" ::: "memory");
#endif
}
#endif

// ---------------------------------------------------------------------------
// Pass 1: pack two f32 matrices [rows x 2048] into one bf16 [rows x 4096]
// dst[row][0:2048) = L[row], dst[row][2048:4096) = R[row]
// ---------------------------------------------------------------------------
__global__ void pack_concat_bf16(const float* __restrict__ L,
                                 const float* __restrict__ R,
                                 __bf16* __restrict__ dst, int rows)
{
    int t = blockIdx.x * blockDim.x + threadIdx.x;
    int total = rows * (K_ / 8);
    if (t >= total) return;
    int row = t / (K_ / 8);
    int col = (t - row * (K_ / 8)) * 8;
    const float* src = (col < IN_) ? (L + (size_t)row * IN_ + col)
                                   : (R + (size_t)row * H_ + (col - IN_));
    float4 a = ((const float4*)src)[0];
    float4 b = ((const float4*)src)[1];
    union { __bf16 h[8]; uint4 q; } u;
    u.h[0] = (__bf16)a.x; u.h[1] = (__bf16)a.y;
    u.h[2] = (__bf16)a.z; u.h[3] = (__bf16)a.w;
    u.h[4] = (__bf16)b.x; u.h[5] = (__bf16)b.y;
    u.h[6] = (__bf16)b.z; u.h[7] = (__bf16)b.w;
    *(uint4*)(dst + (size_t)row * K_ + col) = u.q;
}

// ---------------------------------------------------------------------------
// Pass 2: fused 4-gate WMMA GEMM + LSTM epilogue.
//   Xc : [B_][K_] bf16  (x | h0)
//   Wc : [4][H_][K_] bf16, gate order {g, i, f, o}, row n K-contiguous
// Each wave: 16 columns, 4 gates, 64 rows -> 16 v8f accumulators.
// Double-buffered LDS A-tile; one workgroup barrier per 64-wide K step.
// ---------------------------------------------------------------------------
struct FragBF { union { v16bf v; uint4 q[2]; }; };

__global__ __launch_bounds__(256, 1)
void lstm_wmma_kernel(const __bf16* __restrict__ Xc,
                      const __bf16* __restrict__ Wc,
                      const float*  __restrict__ c0,
                      const float*  __restrict__ bgx, const float* __restrict__ bgh,
                      const float*  __restrict__ bix, const float* __restrict__ bih,
                      const float*  __restrict__ bfx, const float* __restrict__ bfh,
                      const float*  __restrict__ box, const float* __restrict__ boh,
                      float* __restrict__ hy, float* __restrict__ cy)
{
    __shared__ __bf16 ldsA[2][BM][BKP];

    const int tid  = threadIdx.x;
    const int wave = tid >> 5;
    const int lane = tid & 31;
    const int m0    = blockIdx.y * BM;
    const int nbase = blockIdx.x * BN + wave * 16;

    const int ln   = lane & 15;     // row-within-tile (A) / column (B)
    const int hi   = lane >> 4;     // half-wave select
    const int hi8  = hi * 8;        // unused; A also offsets K by hi16
    const int hi16 = hi * 16;       // B-fragment K sub-offset
    const int ncol = nbase + ln;

    v8f acc[4][4];                  // [m-tile][gate]
    #pragma unroll
    for (int mt = 0; mt < 4; ++mt)
        #pragma unroll
        for (int g = 0; g < 4; ++g)
            acc[mt][g] = (v8f)0.0f;

    // A-tile cooperative copy: 256 threads x 32B = 64x64 bf16 per tile
    const int crow = tid >> 2;
    const int ccol = (tid & 3) * 16;
    const __bf16* gsrc = Xc + (size_t)(m0 + crow) * K_ + ccol;

    const __bf16* wbase[4];
    #pragma unroll
    for (int g = 0; g < 4; ++g)
        wbase[g] = Wc + ((size_t)g * H_ + ncol) * K_ + hi16;

    const int NIT = K_ / BK;        // 64 iterations

    // Prologue: tile 0 -> buffer 0
    {
#if USE_ASYNC_LDS
        async_cp16(gsrc,     &ldsA[0][crow][ccol]);
        async_cp16(gsrc + 8, &ldsA[0][crow][ccol + 8]);
        async_wait0();
#else
        uint4 p0 = ((const uint4*)gsrc)[0];
        uint4 p1 = ((const uint4*)gsrc)[1];
        *(uint4*)&ldsA[0][crow][ccol]     = p0;
        *(uint4*)&ldsA[0][crow][ccol + 8] = p1;
#endif
        __syncthreads();
    }

    for (int kt = 0; kt < NIT; ++kt) {
        const int cur = kt & 1;
#if !USE_ASYNC_LDS
        uint4 pf0, pf1;
#endif
        if (kt + 1 < NIT) {
            const __bf16* s = gsrc + (size_t)(kt + 1) * BK;
#if USE_ASYNC_LDS
            async_cp16(s,     &ldsA[1 - cur][crow][ccol]);
            async_cp16(s + 8, &ldsA[1 - cur][crow][ccol + 8]);
#else
            pf0 = ((const uint4*)s)[0];
            pf1 = ((const uint4*)s)[1];
#endif
        }

        #pragma unroll
        for (int s2 = 0; s2 < 2; ++s2) {
            const int kf = kt * BK + s2 * 32;   // global K offset of substep
            FragBF bf[4];
            #pragma unroll
            for (int g = 0; g < 4; ++g) {
                const __bf16* p = wbase[g] + kf;
                bf[g].q[0] = ((const uint4*)p)[0];
                bf[g].q[1] = ((const uint4*)p)[1];
            }
            FragBF af[4];
            #pragma unroll
            for (int mt = 0; mt < 4; ++mt) {
                af[mt].q[0] = *(const uint4*)&ldsA[cur][mt * 16 + ln][s2 * 32 + hi16];
                af[mt].q[1] = *(const uint4*)&ldsA[cur][mt * 16 + ln][s2 * 32 + hi16 + 8];
            }
            #pragma unroll
            for (int mt = 0; mt < 4; ++mt)
                #pragma unroll
                for (int g = 0; g < 4; ++g)
                    acc[mt][g] = __builtin_amdgcn_wmma_f32_16x16x32_bf16(
                        false, af[mt].v, false, bf[g].v,
                        (short)0, acc[mt][g], false, false);
        }

        if (kt + 1 < NIT) {
#if USE_ASYNC_LDS
            async_wait0();
#else
            *(uint4*)&ldsA[1 - cur][crow][ccol]     = pf0;
            *(uint4*)&ldsA[1 - cur][crow][ccol + 8] = pf1;
#endif
        }
        __syncthreads();
    }

    // ---- Fused LSTM epilogue (wave-local: this wave owns all 4 gates) ----
    const float bg  = bgx[ncol] + bgh[ncol];
    const float bi  = bix[ncol] + bih[ncol];
    const float bfv = bfx[ncol] + bfh[ncol];
    const float bo  = box[ncol] + boh[ncol];

    #pragma unroll
    for (int mt = 0; mt < 4; ++mt) {
        #pragma unroll
        for (int r = 0; r < 8; ++r) {
            const int m = m0 + mt * 16 + hi * 8 + r;
            const size_t idx = (size_t)m * H_ + ncol;
            float gv = tanhf(acc[mt][0][r] + bg);
            float iv = 1.0f / (1.0f + __expf(-(acc[mt][1][r] + bi)));
            float fv = 1.0f / (1.0f + __expf(-(acc[mt][2][r] + bfv)));
            float ov = 1.0f / (1.0f + __expf(-(acc[mt][3][r] + bo)));
            float c  = fv * c0[idx] + iv * gv;
            cy[idx] = c;
            hy[idx] = ov * tanhf(c);
        }
    }
}

// ---------------------------------------------------------------------------
extern "C" void kernel_launch(void* const* d_in, const int* in_sizes, int n_in,
                              void* d_out, int out_size, void* d_ws, size_t ws_size,
                              hipStream_t stream) {
    const float* input = (const float*)d_in[0];
    const float* h0    = (const float*)d_in[1];
    const float* c0    = (const float*)d_in[2];
    const float* Wgx = (const float*)d_in[3];  const float* bgx = (const float*)d_in[4];
    const float* Wgh = (const float*)d_in[5];  const float* bgh = (const float*)d_in[6];
    const float* Wix = (const float*)d_in[7];  const float* bix = (const float*)d_in[8];
    const float* Wih = (const float*)d_in[9];  const float* bih = (const float*)d_in[10];
    const float* Wfx = (const float*)d_in[11]; const float* bfx = (const float*)d_in[12];
    const float* Wfh = (const float*)d_in[13]; const float* bfh = (const float*)d_in[14];
    const float* Wox = (const float*)d_in[15]; const float* box_ = (const float*)d_in[16];
    const float* Woh = (const float*)d_in[17]; const float* boh = (const float*)d_in[18];

    __bf16* Xc = (__bf16*)d_ws;                                   // 32 MB
    __bf16* Wc = (__bf16*)((char*)d_ws + (size_t)B_ * K_ * 2);    // 64 MB

    float* hy  = (float*)d_out;
    float* cyp = (float*)d_out + (size_t)B_ * H_;

    // Pass 1: pack activations and weights to bf16, concatenated along K.
    {
        int totalX = B_ * (K_ / 8);
        pack_concat_bf16<<<(totalX + 255) / 256, 256, 0, stream>>>(input, h0, Xc, B_);
        const float* Wx[4] = {Wgx, Wix, Wfx, Wox};
        const float* Wh[4] = {Wgh, Wih, Wfh, Woh};
        int totalW = H_ * (K_ / 8);
        for (int g = 0; g < 4; ++g)
            pack_concat_bf16<<<(totalW + 255) / 256, 256, 0, stream>>>(
                Wx[g], Wh[g], Wc + (size_t)g * H_ * K_, H_);
    }

    // Pass 2: fused WMMA GEMM + LSTM gates.
    dim3 grid(H_ / BN, B_ / BM);   // 16 x 64 workgroups
    lstm_wmma_kernel<<<grid, 256, 0, stream>>>(
        Xc, Wc, c0, bgx, bgh, bix, bih, bfx, bfh, box_, boh, hy, cyp);
}
